// RoIProcessingNetwork_9852654977111
// MI455X (gfx1250) — matrix operand order repaired and checked by reading, not compiled, not run on hardware
//
#include <hip/hip_runtime.h>

typedef __attribute__((ext_vector_type(16))) _Float16 v16h;
typedef __attribute__((ext_vector_type(8)))  _Float16 v8h;
typedef __attribute__((ext_vector_type(8)))  float    v8f;
typedef unsigned int u32;
typedef unsigned long long u64;

#define P_NUM   1024
#define CCH     256
#define FEAT    12544   // 256*7*7
#define HID     1024
#define NCLS    80
#define NCLS1   81
#define NCLSPAD 96
#define NREG    320
#define KNMS    2048
#define NCAND   81920   // 1024*80
#define NSORT   131072  // next pow2

// ---------------------------------------------------------------------------
// RoI-Align (multilevel, SR=2, OUT=7). One thread per (p, c, out-pixel).
// ---------------------------------------------------------------------------
__global__ void roi_align_kernel(const float* __restrict__ prop,
                                 const int* __restrict__ imidx,
                                 const float* __restrict__ f0,
                                 const float* __restrict__ f1,
                                 const float* __restrict__ f2,
                                 const float* __restrict__ f3,
                                 _Float16* __restrict__ X)
{
    int t = blockIdx.x * blockDim.x + threadIdx.x;
    if (t >= P_NUM * CCH * 49) return;
    int pix = t % 49;
    int c   = (t / 49) % CCH;
    int p   = t / (49 * CCH);

    float x1 = prop[4*p+0], y1 = prop[4*p+1], x2 = prop[4*p+2], y2 = prop[4*p+3];
    int b = imidx[p];
    float w = x2 - x1, h = y2 - y1;
    float lv = floorf(4.0f + log2f(sqrtf(w*h) / 224.0f + 1e-8f));
    int lvl = (int)fminf(fmaxf(lv, 2.0f), 5.0f) - 2;
    const float* f; int H, W;
    if      (lvl == 0) { f = f0; H = 160; W = 160; }
    else if (lvl == 1) { f = f1; H = 80;  W = 80;  }
    else if (lvl == 2) { f = f2; H = 40;  W = 40;  }
    else               { f = f3; H = 20;  W = 20;  }
    float stride = (float)(4 << lvl);
    float x1s = x1/stride - 0.5f, y1s = y1/stride - 0.5f;
    float x2s = x2/stride - 0.5f, y2s = y2/stride - 0.5f;
    float bw = (x2s - x1s) * (1.0f/7.0f), bh = (y2s - y1s) * (1.0f/7.0f);
    int oy = pix / 7, ox = pix % 7;
    const float* plane = f + ((size_t)b*CCH + c) * (size_t)(H*W);

    float acc = 0.0f;
    #pragma unroll
    for (int sy = 0; sy < 2; ++sy) {
        float gy = y1s + bh * ((float)oy + ((float)sy + 0.5f) * 0.5f);
        bool vy = (gy > -1.0f) && (gy < (float)H);
        float cy = fminf(fmaxf(gy, 0.0f), (float)H - 1.0f);
        int y0 = (int)floorf(cy);
        int y1i = min(y0 + 1, H - 1);
        float fy = cy - (float)y0;
        #pragma unroll
        for (int sx = 0; sx < 2; ++sx) {
            float gx = x1s + bw * ((float)ox + ((float)sx + 0.5f) * 0.5f);
            bool vx = (gx > -1.0f) && (gx < (float)W);
            float cx = fminf(fmaxf(gx, 0.0f), (float)W - 1.0f);
            int x0 = (int)floorf(cx);
            int x1i = min(x0 + 1, W - 1);
            float fx = cx - (float)x0;
            if (vy && vx) {
                float v00 = plane[y0*W + x0],  v01 = plane[y0*W + x1i];
                float v10 = plane[y1i*W + x0], v11 = plane[y1i*W + x1i];
                acc += (1.0f-fy)*((1.0f-fx)*v00 + fx*v01)
                     +        fy*((1.0f-fx)*v10 + fx*v11);
            }
        }
    }
    X[(size_t)p*FEAT + c*49 + pix] = (_Float16)(acc * 0.25f);
}

// ---------------------------------------------------------------------------
// f32 -> f16 conversion with zero padding (for weight matrices).
// ---------------------------------------------------------------------------
__global__ void cvt_f32_to_f16(const float* __restrict__ src,
                               _Float16* __restrict__ dst,
                               int srcCount, int dstCount)
{
    int t = blockIdx.x * blockDim.x + threadIdx.x;
    if (t >= dstCount) return;
    dst[t] = (t < srcCount) ? (_Float16)src[t] : (_Float16)0.0f;
}

// ---------------------------------------------------------------------------
// GEMM: out[m][n] = sum_k A[m][k] * B[n][k] (+bias, optional relu).
// One wave computes a 32x32 output block (2x2 WMMA tiles): per K-step it
// loads 2 A-fragments + 2 B-fragments and issues 4 v_wmma_f32_16x16x32_f16,
// doubling arithmetic intensity vs 1 tile/wave and giving the scheduler 4
// independent accumulator chains. Fragments hold k contiguously: k = 16*(lane/16) + s.
// Requires M % 32 == 0 and N % 32 == 0.
// ---------------------------------------------------------------------------
__global__ void gemm_wmma_kernel(const _Float16* __restrict__ A,
                                 const _Float16* __restrict__ B,
                                 const float* __restrict__ bias, int biasN,
                                 float* __restrict__ outF,
                                 _Float16* __restrict__ outH,
                                 int M, int N, int K, int relu)
{
    int wave = (int)((blockIdx.x * (u32)blockDim.x + threadIdx.x) >> 5);
    int lane = threadIdx.x & 31;
    int nBlocks = N >> 5;
    int mBlocks = M >> 5;
    if (wave >= mBlocks * nBlocks) return;
    int tm = (wave / nBlocks) << 5;
    int tn = (wave % nBlocks) << 5;
    int l15 = lane & 15;
    int hi  = lane >> 4;

    // A lane row: lanes 0-15 hold K=0..15, lanes 16-31 K=16..31
    const _Float16* arow0 = A + (size_t)(tm + l15) * K + hi * 16;
    const _Float16* arow1 = arow0 + (size_t)16 * K;
    // B lane col: lanes 0-15 hold K=0..15, lanes 16-31 K=16..31 of column l15
    const _Float16* brow0 = B + (size_t)(tn + l15) * K + hi * 16;
    const _Float16* brow1 = brow0 + (size_t)16 * K;

    v8f acc00 = {}, acc01 = {}, acc10 = {}, acc11 = {};
    for (int k0 = 0; k0 < K; k0 += 32) {
        v8h a0lo = *(const v8h*)(arow0 + k0);
        v8h a0hi = *(const v8h*)(arow0 + k0 + 8);
        v8h a1lo = *(const v8h*)(arow1 + k0);
        v8h a1hi = *(const v8h*)(arow1 + k0 + 8);
        v16h a0 = __builtin_shufflevector(a0lo, a0hi,
                      0,1,2,3,4,5,6,7,8,9,10,11,12,13,14,15);
        v16h a1 = __builtin_shufflevector(a1lo, a1hi,
                      0,1,2,3,4,5,6,7,8,9,10,11,12,13,14,15);
        v16h b0 = *(const v16h*)(brow0 + k0);
        v16h b1 = *(const v16h*)(brow1 + k0);
        acc00 = __builtin_amdgcn_wmma_f32_16x16x32_f16(
                    false, a0, false, b0, (short)0, acc00, false, false);
        acc01 = __builtin_amdgcn_wmma_f32_16x16x32_f16(
                    false, a0, false, b1, (short)0, acc01, false, false);
        acc10 = __builtin_amdgcn_wmma_f32_16x16x32_f16(
                    false, a1, false, b0, (short)0, acc10, false, false);
        acc11 = __builtin_amdgcn_wmma_f32_16x16x32_f16(
                    false, a1, false, b1, (short)0, acc11, false, false);
    }

    #pragma unroll
    for (int ti = 0; ti < 2; ++ti) {
        #pragma unroll
        for (int tj = 0; tj < 2; ++tj) {
            v8f acc = (ti == 0) ? (tj == 0 ? acc00 : acc01)
                                : (tj == 0 ? acc10 : acc11);
            int n = tn + tj * 16 + l15;
            float bv = (n < biasN) ? bias[n] : 0.0f;
            #pragma unroll
            for (int r = 0; r < 8; ++r) {
                int m = tm + ti * 16 + r + hi * 8; // C/D: VGPR r -> row r (+8 hi)
                float v = acc[r] + bv;
                if (relu) v = fmaxf(v, 0.0f);
                if (outF) outF[(size_t)m * N + n] = v;
                if (outH) outH[(size_t)m * N + n] = (_Float16)v;
            }
        }
    }
}

// ---------------------------------------------------------------------------
// Softmax prep per proposal (max + 1/sum over 81 logits).
// ---------------------------------------------------------------------------
__global__ void softmax_prep_kernel(const float* __restrict__ logits,
                                    float* __restrict__ mxArr,
                                    float* __restrict__ invArr)
{
    int p = blockIdx.x * blockDim.x + threadIdx.x;
    if (p >= P_NUM) return;
    const float* lg = logits + (size_t)p * NCLSPAD;
    float mx = lg[0];
    for (int c = 1; c < NCLS1; ++c) mx = fmaxf(mx, lg[c]);
    float s = 0.0f;
    for (int c = 0; c < NCLS1; ++c) s += expf(lg[c] - mx);
    mxArr[p] = mx;
    invArr[p] = 1.0f / s;
}

// ---------------------------------------------------------------------------
// Per (proposal, class): softmax score, box decode + clip, validity, sort key.
// key = (bits(score_masked) << 32) | (~idx)  -> descending sort == jax top_k.
// ---------------------------------------------------------------------------
__global__ void decode_kernel(const float* __restrict__ logits,
                              const float* __restrict__ regF,
                              const float* __restrict__ mxArr,
                              const float* __restrict__ invArr,
                              const float* __restrict__ prop,
                              const int* __restrict__ imidx,
                              const float* __restrict__ imsizes,
                              float4* __restrict__ boxes,
                              u64* __restrict__ keys)
{
    int idx = blockIdx.x * blockDim.x + threadIdx.x;
    if (idx >= NCAND) return;
    int p = idx / NCLS, c = idx % NCLS;
    float scr = expf(logits[(size_t)p*NCLSPAD + c] - mxArr[p]) * invArr[p];
    const float* r = regF + (size_t)p*NREG + c*4;
    const float CLIPV = 4.135166556742356f; // log(1000/16)
    float dx = r[0]*0.1f, dy = r[1]*0.1f;
    float dw = fminf(r[2]*0.2f, CLIPV), dh = fminf(r[3]*0.2f, CLIPV);
    float x1 = prop[4*p], y1 = prop[4*p+1], x2 = prop[4*p+2], y2 = prop[4*p+3];
    float pw = x2-x1, ph = y2-y1;
    float pcx = x1 + 0.5f*pw, pcy = y1 + 0.5f*ph;
    float ncx = pcx + dx*pw, ncy = pcy + dy*ph;
    float nw = pw*expf(dw), nh = ph*expf(dh);
    int im = imidx[p];
    float Hm = imsizes[im*2+0], Wm = imsizes[im*2+1];
    float bx1 = fminf(fmaxf(ncx - 0.5f*nw, 0.0f), Wm);
    float by1 = fminf(fmaxf(ncy - 0.5f*nh, 0.0f), Hm);
    float bx2 = fminf(fmaxf(ncx + 0.5f*nw, 0.0f), Wm);
    float by2 = fminf(fmaxf(ncy + 0.5f*nh, 0.0f), Hm);
    bool valid = (scr > 0.05f) && (bx2 - bx1 > 0.0f) && (by2 - by1 > 0.0f);
    float sm = valid ? scr : 0.0f;
    boxes[idx] = make_float4(bx1, by1, bx2, by2);
    keys[idx] = ((u64)__float_as_uint(sm) << 32) | (u64)(0xFFFFFFFFu - (u32)idx);
}

__global__ void pad_keys_kernel(u64* __restrict__ keys)
{
    int t = blockIdx.x * blockDim.x + threadIdx.x;
    int idx = NCAND + t;
    if (idx < NSORT) keys[idx] = 0ull;
}

// ---------------------------------------------------------------------------
// Bitonic sort (descending) of NSORT u64 keys.
// ---------------------------------------------------------------------------
__global__ void bitonic_local_kernel(u64* __restrict__ keys, int fullSort, int gsize)
{
    __shared__ u64 sk[2048];
    int base = blockIdx.x << 11;
    int t = threadIdx.x; // 1024 threads
    sk[t]        = keys[base + t];
    sk[t + 1024] = keys[base + t + 1024];
    __syncthreads();
    if (fullSort) {
        for (int sz = 2; sz <= 2048; sz <<= 1) {
            for (int st = sz >> 1; st >= 1; st >>= 1) {
                int i = ((t & ~(st - 1)) << 1) | (t & (st - 1));
                int j = i | st;
                bool desc = (((base + i) & sz) == 0);
                u64 a = sk[i], b = sk[j];
                bool sw = desc ? (a < b) : (a > b);
                if (sw) { sk[i] = b; sk[j] = a; }
                __syncthreads();
            }
        }
    } else {
        for (int st = 1024; st >= 1; st >>= 1) {
            int i = ((t & ~(st - 1)) << 1) | (t & (st - 1));
            int j = i | st;
            bool desc = (((base + i) & gsize) == 0);
            u64 a = sk[i], b = sk[j];
            bool sw = desc ? (a < b) : (a > b);
            if (sw) { sk[i] = b; sk[j] = a; }
            __syncthreads();
        }
    }
    keys[base + t]        = sk[t];
    keys[base + t + 1024] = sk[t + 1024];
}

__global__ void bitonic_global_kernel(u64* __restrict__ keys, int gsize, int stride)
{
    int t = blockIdx.x * blockDim.x + threadIdx.x;
    if (t >= NSORT / 2) return;
    int i = ((t & ~(stride - 1)) << 1) | (t & (stride - 1));
    int j = i | stride;
    bool desc = ((i & gsize) == 0);
    u64 a = keys[i], b = keys[j];
    bool sw = desc ? (a < b) : (a > b);
    if (sw) { keys[i] = b; keys[j] = a; }
}

// ---------------------------------------------------------------------------
// Gather top-2048 candidates, build group-offset boxes for NMS.
// ---------------------------------------------------------------------------
__global__ void gather_topk_kernel(const u64* __restrict__ keys,
                                   const float4* __restrict__ boxes,
                                   const int* __restrict__ imidx,
                                   float4* __restrict__ bK, float4* __restrict__ obK,
                                   float* __restrict__ sK, float* __restrict__ cK,
                                   int* __restrict__ imK, int* __restrict__ validK)
{
    int t = blockIdx.x * blockDim.x + threadIdx.x;
    if (t >= KNMS) return;
    u64 k = keys[t];
    u32 idx = 0xFFFFFFFFu - (u32)(k & 0xFFFFFFFFull);
    float sv = __uint_as_float((u32)(k >> 32));
    int p = (int)(idx / NCLS), c = (int)(idx % NCLS);
    int im = imidx[p];
    float4 bx = boxes[idx];
    float g = ((float)(im * NCLS + c)) * 100000.0f;
    bK[t]  = bx;
    obK[t] = make_float4(bx.x + g, bx.y + g, bx.z + g, bx.w + g);
    sK[t] = sv;
    cK[t] = (float)c;
    imK[t] = im;
    validK[t] = (sv > 0.0f) ? 1 : 0;
}

// Suppression bitmask: word (row, w) has bit j set iff iou(row, w*32+j) > 0.5
__global__ void supmat_kernel(const float4* __restrict__ ob, u32* __restrict__ sup)
{
    int t = blockIdx.x * blockDim.x + threadIdx.x;
    if (t >= KNMS * 64) return;
    int row = t >> 6, word = t & 63;
    float4 a = ob[row];
    float areaA = (a.z - a.x) * (a.w - a.y);
    u32 bits = 0;
    int j0 = word << 5;
    for (int jj = 0; jj < 32; ++jj) {
        float4 b = ob[j0 + jj];
        float areaB = (b.z - b.x) * (b.w - b.y);
        float ix1 = fmaxf(a.x, b.x), iy1 = fmaxf(a.y, b.y);
        float ix2 = fminf(a.z, b.z), iy2 = fminf(a.w, b.w);
        float iw = fmaxf(ix2 - ix1, 0.0f), ih = fmaxf(iy2 - iy1, 0.0f);
        float inter = iw * ih;
        float iou = inter / (areaA + areaB - inter + 1e-9f);
        if (iou > 0.5f) bits |= (1u << jj);
    }
    sup[t] = bits;
}

// Sequential greedy scan with a 2048-bit removed set (one block, 64 threads).
__global__ void nms_kernel(const u32* __restrict__ sup,
                           const int* __restrict__ validK,
                           float* __restrict__ keepF)
{
    __shared__ u32 removed[64];
    __shared__ int cur;
    int t = threadIdx.x;
    removed[t] = 0u;
    __syncthreads();
    for (int i = 0; i < KNMS; ++i) {
        if (t == 0) {
            int r = (removed[i >> 5] >> (i & 31)) & 1;
            int k = validK[i] && !r;
            cur = k;
            keepF[i] = k ? 1.0f : 0.0f;
        }
        __syncthreads();
        if (cur) removed[t] |= sup[i * 64 + t];
        __syncthreads();
    }
}

// Per-image top-100 by rank-counting over LDS-resident keys.
__global__ void per_image_topk_kernel(const float4* __restrict__ bK,
                                      const float* __restrict__ sK,
                                      const float* __restrict__ cK,
                                      const int* __restrict__ imK,
                                      const float* __restrict__ keepF,
                                      float* __restrict__ out)
{
    __shared__ u64 kk[KNMS];
    int img = blockIdx.x;
    int tid = threadIdx.x;
    for (int e = tid; e < KNMS; e += 256) {
        float s = sK[e] * keepF[e];
        float si = (imK[e] == img) ? s : -1.0f;
        u32 u = __float_as_uint(si);
        u32 mono = (u & 0x80000000u) ? ~u : (u | 0x80000000u);
        kk[e] = ((u64)mono << 32) | (u64)(0xFFFFFFFFu - (u32)e);
    }
    __syncthreads();
    for (int e = tid; e < KNMS; e += 256) {
        u64 mine = kk[e];
        int rank = 0;
        for (int j = 0; j < KNMS; ++j) rank += (kk[j] > mine) ? 1 : 0;
        if (rank < 100) {
            float s = sK[e] * keepF[e];
            float si = (imK[e] == img) ? s : -1.0f;
            float4 b = bK[e];
            int base = (img * 100 + rank) * 4;
            out[base + 0] = b.x; out[base + 1] = b.y;
            out[base + 2] = b.z; out[base + 3] = b.w;
            out[1600 + img * 100 + rank] = fmaxf(si, 0.0f);
            out[2000 + img * 100 + rank] = cK[e];
        }
    }
}

// ---------------------------------------------------------------------------
extern "C" void kernel_launch(void* const* d_in, const int* in_sizes, int n_in,
                              void* d_out, int out_size, void* d_ws, size_t ws_size,
                              hipStream_t stream)
{
    (void)in_sizes; (void)n_in; (void)out_size; (void)ws_size;
    const float* proposals = (const float*)d_in[0];
    const int*   imidx     = (const int*)d_in[1];
    const float* f0        = (const float*)d_in[2];
    const float* f1        = (const float*)d_in[3];
    const float* f2        = (const float*)d_in[4];
    const float* f3        = (const float*)d_in[5];
    const float* imsizes   = (const float*)d_in[6];
    const float* fc0_w     = (const float*)d_in[7];
    const float* fc0_b     = (const float*)d_in[8];
    const float* fc1_w     = (const float*)d_in[9];
    const float* fc1_b     = (const float*)d_in[10];
    const float* cls_w     = (const float*)d_in[11];
    const float* cls_b     = (const float*)d_in[12];
    const float* reg_w     = (const float*)d_in[13];
    const float* reg_b     = (const float*)d_in[14];
    float* out = (float*)d_out;

    char* ws = (char*)d_ws;
    size_t off = 0;
    auto alloc = [&](size_t bytes) -> void* {
        void* p = ws + off;
        off = (off + bytes + 255) & ~(size_t)255;
        return p;
    };

    _Float16* X    = (_Float16*)alloc((size_t)P_NUM * FEAT * 2);
    _Float16* W0h  = (_Float16*)alloc((size_t)HID * FEAT * 2);
    _Float16* W1h  = (_Float16*)alloc((size_t)HID * HID * 2);
    _Float16* Wch  = (_Float16*)alloc((size_t)NCLSPAD * HID * 2);
    _Float16* Wrh  = (_Float16*)alloc((size_t)NREG * HID * 2);
    _Float16* Y0h  = (_Float16*)alloc((size_t)P_NUM * HID * 2);
    _Float16* Y1h  = (_Float16*)alloc((size_t)P_NUM * HID * 2);
    float*    logitsF = (float*)alloc((size_t)P_NUM * NCLSPAD * 4);
    float*    regF    = (float*)alloc((size_t)P_NUM * NREG * 4);
    float*    mxArr   = (float*)alloc(P_NUM * 4);
    float*    invArr  = (float*)alloc(P_NUM * 4);
    float4*   boxes   = (float4*)alloc((size_t)NCAND * 16);
    u64*      keys    = (u64*)alloc((size_t)NSORT * 8);
    float4*   bK      = (float4*)alloc(KNMS * 16);
    float4*   obK     = (float4*)alloc(KNMS * 16);
    float*    sK      = (float*)alloc(KNMS * 4);
    float*    cK      = (float*)alloc(KNMS * 4);
    int*      imK     = (int*)alloc(KNMS * 4);
    int*      validK  = (int*)alloc(KNMS * 4);
    u32*      sup     = (u32*)alloc((size_t)KNMS * 64 * 4);
    float*    keepF   = (float*)alloc(KNMS * 4);

    // Weight conversions to f16 (zero-padded for cls head 81->96)
    {
        int n = HID * FEAT;
        cvt_f32_to_f16<<<(n + 255) / 256, 256, 0, stream>>>(fc0_w, W0h, n, n);
        n = HID * HID;
        cvt_f32_to_f16<<<(n + 255) / 256, 256, 0, stream>>>(fc1_w, W1h, n, n);
        cvt_f32_to_f16<<<(NCLSPAD * HID + 255) / 256, 256, 0, stream>>>(
            cls_w, Wch, NCLS1 * HID, NCLSPAD * HID);
        n = NREG * HID;
        cvt_f32_to_f16<<<(n + 255) / 256, 256, 0, stream>>>(reg_w, Wrh, n, n);
    }

    // RoI-Align -> X (f16, [1024 x 12544])
    {
        int total = P_NUM * CCH * 49;
        roi_align_kernel<<<(total + 255) / 256, 256, 0, stream>>>(
            proposals, imidx, f0, f1, f2, f3, X);
    }

    // fc0: [1024x12544] x [12544x1024]^T -> relu -> Y0h
    {
        int waves = (P_NUM / 32) * (HID / 32);
        gemm_wmma_kernel<<<(waves * 32 + 255) / 256, 256, 0, stream>>>(
            X, W0h, fc0_b, HID, (float*)nullptr, Y0h, P_NUM, HID, FEAT, 1);
    }
    // fc1
    {
        int waves = (P_NUM / 32) * (HID / 32);
        gemm_wmma_kernel<<<(waves * 32 + 255) / 256, 256, 0, stream>>>(
            Y0h, W1h, fc1_b, HID, (float*)nullptr, Y1h, P_NUM, HID, HID, 1);
    }
    // cls head (padded N=96)
    {
        int waves = (P_NUM / 32) * (NCLSPAD / 32);
        gemm_wmma_kernel<<<(waves * 32 + 255) / 256, 256, 0, stream>>>(
            Y1h, Wch, cls_b, NCLS1, logitsF, (_Float16*)nullptr, P_NUM, NCLSPAD, HID, 0);
    }
    // reg head (N=320)
    {
        int waves = (P_NUM / 32) * (NREG / 32);
        gemm_wmma_kernel<<<(waves * 32 + 255) / 256, 256, 0, stream>>>(
            Y1h, Wrh, reg_b, NREG, regF, (_Float16*)nullptr, P_NUM, NREG, HID, 0);
    }

    softmax_prep_kernel<<<(P_NUM + 255) / 256, 256, 0, stream>>>(logitsF, mxArr, invArr);
    decode_kernel<<<(NCAND + 255) / 256, 256, 0, stream>>>(
        logitsF, regF, mxArr, invArr, proposals, imidx, imsizes, boxes, keys);
    pad_keys_kernel<<<((NSORT - NCAND) + 255) / 256, 256, 0, stream>>>(keys);

    // Bitonic sort (descending) of 131072 keys
    bitonic_local_kernel<<<NSORT / 2048, 1024, 0, stream>>>(keys, 1, 0);
    for (int gsize = 4096; gsize <= NSORT; gsize <<= 1) {
        for (int stride = gsize >> 1; stride >= 2048; stride >>= 1)
            bitonic_global_kernel<<<(NSORT / 2 + 255) / 256, 256, 0, stream>>>(
                keys, gsize, stride);
        bitonic_local_kernel<<<NSORT / 2048, 1024, 0, stream>>>(keys, 0, gsize);
    }

    gather_topk_kernel<<<(KNMS + 255) / 256, 256, 0, stream>>>(
        keys, boxes, imidx, bK, obK, sK, cK, imK, validK);
    supmat_kernel<<<(KNMS * 64 + 255) / 256, 256, 0, stream>>>(obK, sup);
    nms_kernel<<<1, 64, 0, stream>>>(sup, validK, keepF);
    per_image_topk_kernel<<<4, 256, 0, stream>>>(bK, sK, cK, imK, keepF, out);
}
